// RegionProposalNetwork_14448269983855
// MI455X (gfx1250) — compile-verified, matrix operand order checked by reading, not run on hardware
//
#include <hip/hip_runtime.h>
#include <hip/hip_bf16.h>
#include <cstdint>
#include <cstddef>

// ---------------------------------------------------------------------------
// RPN for MI455X (gfx1250): 3x3 conv + heads via v_wmma_f32_16x16x32_f16.
// Weights pre-converted to f16 blocked layouts so per-chunk staging is a
// single Tensor Data Mover descriptor (tensor_load_to_lds + s_wait_tensorcnt),
// overlapped with compute / VALU halo staging.  Then prep/sort/NMS.
// ---------------------------------------------------------------------------

typedef __attribute__((ext_vector_type(16))) _Float16 v16h;
typedef __attribute__((ext_vector_type(8)))  float    v8f;
typedef __attribute__((ext_vector_type(4)))  unsigned v4u;

union Frag  { unsigned u[8]; v16h h; };
union Pack8 { _Float16 hh[8]; v4u u4; };

// ---- output layout (flat float/int32 concat, reference return order) ------
#define OFF_LOCS   0         // (8,36864,4)  = 1,179,648
#define OFF_SCORES 1179648   // (8,36864,2)  =   589,824
#define OFF_ROIS   1769472   // (2400,4)     =     9,600
#define OFF_RIDX   1779072   // (2400,) int  =     2,400
#define OFF_ANCHOR 1781472   // (36864,4)    =   147,456
#define OFF_NVALID 1928928   // (8,)    int

// ---- workspace layout (bytes) ---------------------------------------------
#define WS_H     0u          // 8*4096*512 f16          = 33,554,432
#define WS_FG    33554432u   // 8*36864 f32             =  1,179,648
#define WS_BOXES 34734080u   // 8*36864*4 f32           =  4,718,592
#define WS_KEYS  39452672u   // 8*65536 u64             =  4,194,304
#define WS_KEPT  43646976u   // 8*6000*4 f32            =    768,000
#define WS_WT    44414976u   // w1 f16 blocked          =  4,718,592
#define WS_WHT   49133568u   // head weights f16        =     65,536

__device__ __forceinline__ int kb(int v, int hi) {
    // 16-bit WMMA fragment: VGPR v holds the K pair starting at this index.
    return 8 * hi + 2 * v + (v >= 4 ? 8 : 0);
}

__device__ __forceinline__ v8f wmma16(const Frag& a, const Frag& b, v8f c) {
    return __builtin_amdgcn_wmma_f32_16x16x32_f16(
        /*neg_a=*/false, a.h, /*neg_b=*/false, b.h,
        /*c_mod=*/(short)0, c, /*reuse_a=*/false, /*reuse_b=*/false);
}

// ---- Tensor Data Mover helper (guarded; sync-copy fallback elsewhere) -----
#if __has_builtin(__builtin_amdgcn_tensor_load_to_lds) && \
    __has_builtin(__builtin_amdgcn_s_wait_tensorcnt)
#define USE_TDM 1
typedef __attribute__((ext_vector_type(4))) unsigned tdm_u4;
typedef __attribute__((ext_vector_type(8))) int      tdm_i8;
typedef __attribute__((ext_vector_type(4))) int      tdm_i4;

// 2D tile load: dim1 rows of dim0 8-byte elements, row stride stride0 (8B
// units), dense into LDS at lds_off.  D# packed per CDNA5 ISA.
__device__ __forceinline__ void tdm_load_2d(const void* gaddr, unsigned lds_off,
                                            unsigned dim0, unsigned dim1,
                                            unsigned stride0) {
    unsigned long long ga = (unsigned long long)gaddr;
    tdm_u4 g0;
    g0[0] = 1u;                                                // count=1, user
    g0[1] = lds_off;                                           // lds_addr
    g0[2] = (unsigned)ga;                                      // addr[31:0]
    g0[3] = (unsigned)((ga >> 32) & 0x01FFFFFFu) | (2u << 30); // addr[56:32]|type=2
    tdm_i8 g1 = {};
    g1[0] = (int)(3u << 16);                           // mask=0, data_size=8B
    g1[1] = (int)(dim0 << 16);                         // tensor_dim0[15:0]
    g1[2] = (int)((dim0 >> 16) | (dim1 << 16));        // dim0 hi | tensor_dim1 lo
    g1[3] = (int)((dim1 >> 16) | (dim0 << 16));        // dim1 hi | tile_dim0
    g1[4] = (int)dim1;                                 // tile_dim1 (tile_dim2=0)
    g1[5] = (int)stride0;                              // tensor_dim0_stride lo32
    g1[6] = 0; g1[7] = 0;
    tdm_i4 gz = {0, 0, 0, 0};
#if defined(__clang_major__) && (__clang_major__ >= 23)
    tdm_i8 g4 = {};
    __builtin_amdgcn_tensor_load_to_lds(g0, g1, gz, gz, g4, 0);
#else
    __builtin_amdgcn_tensor_load_to_lds(g0, g1, gz, gz, 0);
#endif
}

__device__ __forceinline__ unsigned lds_off_of(const void* p) {
    return (unsigned)(unsigned long long)p;   // generic LDS ptr: low 32 = offset
}
#endif

// ===========================================================================
// Kernel 0a: pre-convert conv weights to f16 blocked [ci/32][tap][co][ci%32]
// so that one (chunk, coTile) slab is a 9-row x 4KB, 32KB-stride 2D tile.
// ===========================================================================
__global__ __launch_bounds__(256)
void k_cvtw(const float* __restrict__ w1, _Float16* __restrict__ wT) {
    int idx = blockIdx.x * 256 + threadIdx.x;          // 512*512*9 = 2,359,296
    int co = idx / 4608;
    int r  = idx - co * 4608;
    int ci = r / 9;
    int kp = r - ci * 9;
    wT[(((size_t)(ci >> 5) * 9 + kp) * 512 + co) * 32 + (ci & 31)] =
        (_Float16)w1[idx];
}

// ===========================================================================
// Kernel 0b: pre-convert head weights (36 loc + 18 score, pad to 64) to f16
// blocked [ci/32][c][ci%32]: each K-chunk is one contiguous 4KB slab.
// ===========================================================================
__global__ __launch_bounds__(256)
void k_cvthw(const float* __restrict__ w_loc, const float* __restrict__ w_score,
             _Float16* __restrict__ wHT) {
    int idx = blockIdx.x * 256 + threadIdx.x;          // 64*512 = 32,768
    int c  = idx >> 9;
    int ci = idx & 511;
    float v = 0.0f;
    if (c < 36)      v = w_loc[c * 512 + ci];
    else if (c < 54) v = w_score[(c - 36) * 512 + ci];
    wHT[((size_t)(ci >> 5) * 64 + c) * 32 + (ci & 31)] = (_Float16)v;
}

// ===========================================================================
// Kernel 1: 3x3 conv (512->512) + bias + ReLU.  h stored [n][pixel][co] f16.
// Block: 256 threads = 8 waves; tile 64 px (one row) x 64 co, 2 C-tiles/wave.
// Weights DMA'd by TDM while input halo is staged (zero-padded) by VALU.
// ===========================================================================
__global__ __launch_bounds__(256)
void k_conv(const float* __restrict__ xin, const _Float16* __restrict__ wT,
            const float* __restrict__ b1, _Float16* __restrict__ h) {
    __shared__ _Float16 lw[9][64][32];   // [tap][co][ci] 36.0 KB
    __shared__ _Float16 lx[3][66][32];   // [dy][x+1][ci] 12.4 KB

    const int b      = blockIdx.x;
    const int coTile = (b & 7) * 64;
    const int y      = (b >> 3) & 63;
    const int n      = b >> 9;
    const int tid  = threadIdx.x;
    const int w    = tid >> 5;
    const int l    = tid & 31;
    const int pxB  = (w & 3) * 16;
    const int coB0 = (w >> 2) * 32;
    const int m    = l & 15;
    const int hi   = l >> 4;

    v8f acc0 = {}; v8f acc1 = {};

    for (int cc = 0; cc < 512; cc += 32) {
        __syncthreads();   // prior tap-loop done reading lw/lx
        const _Float16* wsrc =
            wT + ((size_t)(cc >> 5) * 9 * 512 + coTile) * 32;
#ifdef USE_TDM
        if (tid < 32)      // one 2D descriptor: 9 rows x 4KB, 32KB stride
            tdm_load_2d(wsrc, lds_off_of(&lw[0][0][0]), 512u, 9u, 4096u);
#else
        {
            v4u* dst = (v4u*)&lw[0][0][0];
            for (int idx = tid; idx < 2304; idx += 256) {
                int kp = idx >> 8, off = idx & 255;
                dst[idx] = ((const v4u*)(wsrc + (size_t)kp * 512 * 32))[off];
            }
        }
#endif
        // ---- stage 3 halo rows of input (zero-padded): overlaps the DMA ----
        for (int idx = tid; idx < 6336; idx += 256) {
            int dyc = idx / 66;
            int xx  = idx - dyc * 66;
            int dy  = dyc >> 5;
            int ci  = dyc & 31;
            int gy  = y + dy - 1, gx = xx - 1;
            float v = 0.0f;
            if ((unsigned)gy < 64u && (unsigned)gx < 64u)
                v = xin[(size_t)(n * 512 + cc + ci) * 4096 + gy * 64 + gx];
            lx[dy][xx][ci] = (_Float16)v;
        }
#ifdef USE_TDM
        if (tid < 32) __builtin_amdgcn_s_wait_tensorcnt(0);
#endif
        __syncthreads();

        #pragma unroll
        for (int kp = 0; kp < 9; ++kp) {
            const int dy = kp / 3, dx = kp - dy * 3;
            Frag fa0, fa1, fb;
            #pragma unroll
            for (int v = 0; v < 8; ++v) {
                int K = kb(v, hi);
                fb.u[v]  = *(const unsigned*)&lx[dy][pxB + m + dx][K];
                fa0.u[v] = *(const unsigned*)&lw[kp][coB0 + m][K];
                fa1.u[v] = *(const unsigned*)&lw[kp][coB0 + 16 + m][K];
            }
            acc0 = wmma16(fa0, fb, acc0);
            acc1 = wmma16(fa1, fb, acc1);
        }
    }

    // ---- epilogue: bias + ReLU + pack 8 f16 -> b128 store ----
    const int px = pxB + m;
    const int p  = y * 64 + px;
    #pragma unroll
    for (int t = 0; t < 2; ++t) {
        v8f acc = t ? acc1 : acc0;
        int coG = coTile + coB0 + t * 16 + 8 * hi;
        Pack8 pk;
        #pragma unroll
        for (int v = 0; v < 8; ++v) {
            float f = acc[v] + b1[coG + v];
            pk.hh[v] = (_Float16)fmaxf(f, 0.0f);
        }
        *(v4u*)&h[(size_t)(n * 4096 + p) * 512 + coG] = pk.u4;
    }
}

// ===========================================================================
// Kernel 2: fused 1x1 heads (loc 36 + score 18, padded to 64) via WMMA.
// h row tile staged once by TDM; weight chunks double-buffered by TDM with
// the next chunk's DMA issued before this chunk's WMMA.
// ===========================================================================
__global__ __launch_bounds__(256)
void k_heads(const _Float16* __restrict__ h, const _Float16* __restrict__ wHT,
             const float* __restrict__ b_loc, const float* __restrict__ b_score,
             float* __restrict__ out, float* __restrict__ fg) {
    __shared__ _Float16 lh[32][512];     // 32 KB activation tile
    __shared__ _Float16 lhw[2][64][32];  // 2 x 4 KB weight chunks

    const int b    = blockIdx.x;
    const int half = b & 127;
    const int n    = b >> 7;
    const int p0   = half * 32;
    const int tid  = threadIdx.x;
    const int w    = tid >> 5;
    const int l    = tid & 31;
    const int pxB  = (w & 1) * 16;
    const int coB  = (w >> 1) * 16;
    const int m    = l & 15;
    const int hi   = l >> 4;

    const _Float16* hsrc = h + (size_t)(n * 4096 + p0) * 512;

#ifdef USE_TDM
    if (tid < 32) {
        tdm_load_2d(hsrc, lds_off_of(&lh[0][0]), 4096u, 1u, 4096u);   // 32 KB
        tdm_load_2d(wHT, lds_off_of(&lhw[0][0][0]), 512u, 1u, 512u);  //  4 KB
        __builtin_amdgcn_s_wait_tensorcnt(0);
    }
    __syncthreads();
#else
    {
        const v4u* src = (const v4u*)hsrc;
        v4u* dst = (v4u*)&lh[0][0];
        for (int t = tid; t < 2048; t += 256) dst[t] = src[t];
        v4u* dw = (v4u*)&lhw[0][0][0];
        for (int t = tid; t < 256; t += 256) dw[t] = ((const v4u*)wHT)[t];
        __syncthreads();
    }
#endif

    v8f acc = {};
    for (int cc16 = 0; cc16 < 16; ++cc16) {
        const int cur = cc16 & 1;
        // prefetch next weight chunk into the other buffer
        if (cc16 < 15) {
            const _Float16* nw = wHT + (size_t)(cc16 + 1) * 2048;
#ifdef USE_TDM
            if (tid < 32)
                tdm_load_2d(nw, lds_off_of(&lhw[cur ^ 1][0][0]), 512u, 1u, 512u);
#else
            v4u* dw = (v4u*)&lhw[cur ^ 1][0][0];
            for (int t = tid; t < 256; t += 256) dw[t] = ((const v4u*)nw)[t];
#endif
        }
        Frag fa, fb;
        #pragma unroll
        for (int v = 0; v < 8; ++v) {
            int K = kb(v, hi);
            fb.u[v] = *(const unsigned*)&lh[pxB + m][cc16 * 32 + K];
            fa.u[v] = *(const unsigned*)&lhw[cur][coB + m][K];
        }
        acc = wmma16(fa, fb, acc);
#ifdef USE_TDM
        if (tid < 32) __builtin_amdgcn_s_wait_tensorcnt(0);
#endif
        __syncthreads();
    }

    // ---- epilogue: scatter locs/scores in reference NHWC-flat order ----
    const int px = pxB + m;
    const int p  = p0 + px;
    const int c0 = coB + 8 * hi;
    float vals[8];
    #pragma unroll
    for (int v = 0; v < 8; ++v) {
        int c = c0 + v;
        float f = acc[v];
        if (c < 36)      f += b_loc[c];
        else if (c < 54) f += b_score[c - 36];
        vals[v] = f;
        if (c < 36)
            out[OFF_LOCS + (size_t)n * 147456 + p * 36 + c] = f;
        else if (c < 54)
            out[OFF_SCORES + (size_t)n * 73728 + p * 18 + (c - 36)] = f;
    }
    // softmax fg score: pairs (2a, 2a+1) sit in adjacent regs of one lane
    #pragma unroll
    for (int v = 0; v < 8; v += 2) {
        int c = c0 + v;
        if (c >= 36 && c + 1 < 54) {
            int a = (c - 36) >> 1;
            float mx = fmaxf(vals[v], vals[v + 1]);
            float e0 = __expf(vals[v] - mx), e1 = __expf(vals[v + 1] - mx);
            fg[(size_t)n * 36864 + p * 9 + a] = e1 / (e0 + e1);
        }
    }
}

// ===========================================================================
// Kernel 3: anchors + loc2bbox + clip + min-size + sort keys.
// ===========================================================================
__global__ __launch_bounds__(256)
void k_prep(float* __restrict__ out, const float* __restrict__ fg,
            float* __restrict__ boxes, unsigned long long* __restrict__ keys,
            const int* __restrict__ imh_p, const int* __restrict__ imw_p) {
    const int gid = blockIdx.x * 256 + threadIdx.x;
    const int n = gid >> 16;
    const unsigned i = (unsigned)gid & 65535u;
    if (i >= 36864u) { keys[(size_t)n * 65536 + i] = 0ull; return; }

    const int p = i / 9, a = i - p * 9;
    const int ay = p >> 6, ax = p & 63;
    const float ratios[3] = {0.5f, 1.0f, 2.0f};
    const float scales[3] = {8.0f, 16.0f, 32.0f};
    const int ri = a / 3, si = a - ri * 3;
    const float ah = 16.0f * scales[si] * sqrtf(ratios[ri]);
    const float aw = 16.0f * scales[si] * sqrtf(1.0f / ratios[ri]);
    const float cy = ay * 16.0f + 8.0f, cx = ax * 16.0f + 8.0f;
    const float ay1 = cy - 0.5f * ah, ax1 = cx - 0.5f * aw;
    const float ay2 = cy + 0.5f * ah, ax2 = cx + 0.5f * aw;
    if (n == 0) {
        out[OFF_ANCHOR + (size_t)i * 4 + 0] = ay1;
        out[OFF_ANCHOR + (size_t)i * 4 + 1] = ax1;
        out[OFF_ANCHOR + (size_t)i * 4 + 2] = ay2;
        out[OFF_ANCHOR + (size_t)i * 4 + 3] = ax2;
    }

    const size_t lb = OFF_LOCS + (size_t)n * 147456 + (size_t)i * 4;
    const float dy = out[lb + 0], dx = out[lb + 1];
    const float dh = out[lb + 2], dw = out[lb + 3];
    const float h0 = ay2 - ay1, w0 = ax2 - ax1;
    const float cy2 = dy * h0 + (ay1 + 0.5f * h0);
    const float cx2 = dx * w0 + (ax1 + 0.5f * w0);
    const float h2 = __expf(dh) * h0, w2 = __expf(dw) * w0;
    const float IH = (float)imh_p[0], IW = (float)imw_p[0];
    const float y1 = fminf(fmaxf(cy2 - 0.5f * h2, 0.0f), IH);
    const float x1 = fminf(fmaxf(cx2 - 0.5f * w2, 0.0f), IW);
    const float y2 = fminf(fmaxf(cy2 + 0.5f * h2, 0.0f), IH);
    const float x2 = fminf(fmaxf(cx2 + 0.5f * w2, 0.0f), IW);

    const size_t bb = ((size_t)n * 36864 + i) * 4;
    boxes[bb + 0] = y1; boxes[bb + 1] = x1;
    boxes[bb + 2] = y2; boxes[bb + 3] = x2;

    const bool valid = ((y2 - y1) >= 16.0f) && ((x2 - x1) >= 16.0f);
    const float s = valid ? fg[(size_t)n * 36864 + i] : -__builtin_inff();
    unsigned sb = __float_as_uint(s);
    sb ^= (sb & 0x80000000u) ? 0xFFFFFFFFu : 0x80000000u;
    keys[(size_t)n * 65536 + i] = ((unsigned long long)sb << 32) | i;
}

// ===========================================================================
// Kernel 4: per-image global bitonic sort of 65536 u64 keys, descending.
// ===========================================================================
__global__ __launch_bounds__(1024)
void k_sort(unsigned long long* __restrict__ keys) {
    unsigned long long* k0 = keys + (size_t)blockIdx.x * 65536;
    for (unsigned kk = 2; kk <= 65536u; kk <<= 1) {
        for (unsigned j = kk >> 1; j > 0; j >>= 1) {
            for (unsigned idx = threadIdx.x; idx < 65536u; idx += 1024) {
                unsigned lx = idx ^ j;
                if (lx > idx) {
                    unsigned long long A = k0[idx], B = k0[lx];
                    bool sw = ((idx & kk) == 0) ? (A < B) : (A > B);
                    if (sw) { k0[idx] = B; k0[lx] = A; }
                }
            }
            __syncthreads();
        }
    }
}

// ===========================================================================
// Kernel 5: greedy NMS, one wave per image; emits rois / roi_indices / n_valid.
// ===========================================================================
__global__ __launch_bounds__(32)
void k_nms(const unsigned long long* __restrict__ keys,
           const float* __restrict__ boxes, float* __restrict__ keptg,
           float* __restrict__ out) {
    const int n = blockIdx.x;
    const int l = threadIdx.x;
    for (int t = l; t < 1200; t += 32) out[OFF_ROIS + n * 1200 + t] = 0.0f;
    for (int t = l; t < 300; t += 32)
        reinterpret_cast<int*>(out)[OFF_RIDX + n * 300 + t] = n;

    __shared__ int s_flag;
    const unsigned long long* kk = keys + (size_t)n * 65536;
    float* kept = keptg + (size_t)n * 6000 * 4;
    int nk = 0;

    for (int c = 0; c < 6000; ++c) {
        const unsigned long long key = kk[c];             // uniform across wave
        const unsigned sortable = (unsigned)(key >> 32);
        if (sortable < 0x80000000u) continue;             // -inf / padding
        const unsigned idx = (unsigned)key;
        const size_t bb = ((size_t)n * 36864 + idx) * 4;
        const float by1 = boxes[bb + 0], bx1 = boxes[bb + 1];
        const float by2 = boxes[bb + 2], bx2 = boxes[bb + 3];
        const float areaC = (by2 - by1) * (bx2 - bx1);

        if (l == 0) s_flag = 0;
        __syncthreads();
        int over = 0;
        for (int t = l; t < nk; t += 32) {
            float ky1 = kept[t * 4 + 0], kx1 = kept[t * 4 + 1];
            float ky2 = kept[t * 4 + 2], kx2 = kept[t * 4 + 3];
            float ty = fmaxf(by1, ky1), tx = fmaxf(bx1, kx1);
            float bA = fminf(by2, ky2), bB = fminf(bx2, kx2);
            float ih = fmaxf(bA - ty, 0.0f), iw = fmaxf(bB - tx, 0.0f);
            float inter = ih * iw;
            float areaK = (ky2 - ky1) * (kx2 - kx1);
            if (inter / (areaC + areaK - inter) > 0.7f) over = 1;
        }
        if (over) atomicOr(&s_flag, 1);
        __syncthreads();
        if (s_flag == 0) {
            if (l == 0) {
                kept[nk * 4 + 0] = by1; kept[nk * 4 + 1] = bx1;
                kept[nk * 4 + 2] = by2; kept[nk * 4 + 3] = bx2;
            }
            __threadfence();
            if (nk < 300 && l < 4)
                out[OFF_ROIS + (n * 300 + nk) * 4 + l] =
                    (l == 0) ? by1 : (l == 1) ? bx1 : (l == 2) ? by2 : bx2;
            ++nk;
        }
        __syncthreads();
    }
    if (l == 0)
        reinterpret_cast<int*>(out)[OFF_NVALID + n] = (nk < 300) ? nk : 300;
}

// ===========================================================================
extern "C" void kernel_launch(void* const* d_in, const int* in_sizes, int n_in,
                              void* d_out, int out_size, void* d_ws, size_t ws_size,
                              hipStream_t stream) {
    const float* x       = (const float*)d_in[0];
    const float* w1      = (const float*)d_in[1];
    const float* b1      = (const float*)d_in[2];
    const float* w_score = (const float*)d_in[3];
    const float* b_score = (const float*)d_in[4];
    const float* w_loc   = (const float*)d_in[5];
    const float* b_loc   = (const float*)d_in[6];
    const int*   imh     = (const int*)d_in[7];
    const int*   imw     = (const int*)d_in[8];
    float* out = (float*)d_out;

    char* ws = (char*)d_ws;   // requires ~49.2 MB of scratch
    _Float16* h            = (_Float16*)(ws + WS_H);
    float* fg              = (float*)(ws + WS_FG);
    float* boxes           = (float*)(ws + WS_BOXES);
    unsigned long long* ky = (unsigned long long*)(ws + WS_KEYS);
    float* kept            = (float*)(ws + WS_KEPT);
    _Float16* wT           = (_Float16*)(ws + WS_WT);
    _Float16* wHT          = (_Float16*)(ws + WS_WHT);

    k_cvtw <<<9216, 256, 0, stream>>>(w1, wT);
    k_cvthw<<<128,  256, 0, stream>>>(w_loc, w_score, wHT);
    k_conv <<<4096, 256, 0, stream>>>(x, wT, b1, h);
    k_heads<<<1024, 256, 0, stream>>>(h, wHT, b_loc, b_score, out, fg);
    k_prep <<<2048, 256, 0, stream>>>(out, fg, boxes, ky, imh, imw);
    k_sort <<<8, 1024, 0, stream>>>(ky);
    k_nms  <<<8, 32, 0, stream>>>(ky, boxes, kept, out);
}
